// Attention4D_5669356834809
// MI455X (gfx1250) — compile-verified, matrix operand order checked by reading, not run on hardware
//
#include <hip/hip_runtime.h>
#include <hip/hip_bf16.h>

#define BB   16
#define DIMX 384
#define RRX  28
#define NNX  784
#define NHX  8
#define KDX  32
#define DDX  128
#define DHX  1024
#define HKX  256
#define NPAD 800
#define EPSX 1e-5f
#define SCALEX 0.17677669529663687f

typedef _Float16 v16h __attribute__((ext_vector_type(16)));
typedef _Float16 v8h  __attribute__((ext_vector_type(8)));
typedef float    v8f  __attribute__((ext_vector_type(8)));

// Load a 16x32 f16 WMMA fragment (A layout; B uses same pattern when stored
// as [n][k] row-major). lane<16: row=lane, K 0..7 & 16..23; lane>=16: row=lane-16,
// K 8..15 & 24..31.
static __device__ inline v16h load_frag(const _Float16* base, int stride, int lane, int k0) {
  int r  = lane & 15;
  int ks = (lane >> 4) << 3;
  const _Float16* p = base + r * stride + k0 + ks;
  v8h lo = *(const v8h*)(p);
  v8h hi = *(const v8h*)(p + 16);
  v16h o;
#pragma unroll
  for (int i = 0; i < 8; ++i) { o[i] = lo[i]; o[8 + i] = hi[i]; }
  return o;
}

static __device__ inline v8f wmma16(v16h a, v16h b, v8f c) {
  return __builtin_amdgcn_wmma_f32_16x16x32_f16(false, a, false, b, (short)0, c, false, false);
}

static __device__ inline v8f zero8() {
  v8f c;
#pragma unroll
  for (int i = 0; i < 8; ++i) c[i] = 0.f;
  return c;
}

// ---------------- kernel 0: fold BN into weights, premix th1 bias table ------
__global__ __launch_bounds__(256) void prep_params(
    const float* q_w, const float* q_b, const float* q_g, const float* q_be, const float* q_m, const float* q_rv,
    const float* k_w, const float* k_b, const float* k_g, const float* k_be, const float* k_m, const float* k_rv,
    const float* v_w, const float* v_b, const float* v_g, const float* v_be, const float* v_m, const float* v_rv,
    const float* p_w, const float* p_b, const float* p_g, const float* p_be, const float* p_m, const float* p_rv,
    const float* th1_w, const float* bias_tab,
    _Float16* Wh, float* biasF, _Float16* Wph, float* biasP, float* tabMix) {
  int tid = blockIdx.x * 256 + threadIdx.x;
  if (tid < 1536 * DIMX) {           // fold qkv weights -> f16
    int o = tid / DIMX, i = tid % DIMX;
    const float *w, *g, *rv; int oo;
    if (o < 256)       { w = q_w; g = q_g; rv = q_rv; oo = o; }
    else if (o < 512)  { w = k_w; g = k_g; rv = k_rv; oo = o - 256; }
    else               { w = v_w; g = v_g; rv = v_rv; oo = o - 512; }
    float inv = g[oo] * rsqrtf(rv[oo] + EPSX);
    Wh[tid] = (_Float16)(w[oo * DIMX + i] * inv);
    return;
  }
  tid -= 1536 * DIMX;
  if (tid < DIMX * DHX) {            // fold proj weight -> f16
    int o = tid / DHX;
    float inv = p_g[o] * rsqrtf(p_rv[o] + EPSX);
    Wph[tid] = (_Float16)(p_w[tid] * inv);
    return;
  }
  tid -= DIMX * DHX;
  if (tid < 1536) {                  // fused qkv bias
    int o = tid;
    const float *b, *g, *be, *m, *rv; int oo;
    if (o < 256)       { b = q_b; g = q_g; be = q_be; m = q_m; rv = q_rv; oo = o; }
    else if (o < 512)  { b = k_b; g = k_g; be = k_be; m = k_m; rv = k_rv; oo = o - 256; }
    else               { b = v_b; g = v_g; be = v_be; m = v_m; rv = v_rv; oo = o - 512; }
    float inv = g[oo] * rsqrtf(rv[oo] + EPSX);
    biasF[o] = b[oo] * inv + be[oo] - m[oo] * inv;
    return;
  }
  tid -= 1536;
  if (tid < DIMX) {                  // fused proj bias
    float inv = p_g[tid] * rsqrtf(p_rv[tid] + EPSX);
    biasP[tid] = p_b[tid] * inv + p_be[tid] - p_m[tid] * inv;
    return;
  }
  tid -= DIMX;
  if (tid < NHX * NNX) {             // tabMix = th1_w @ bias_tab
    int h = tid / NNX, t = tid % NNX;
    float acc = 0.f;
#pragma unroll
    for (int h2 = 0; h2 < NHX; ++h2) acc += th1_w[h * NHX + h2] * bias_tab[h2 * NNX + t];
    tabMix[tid] = acc;
  }
}

// ---------------- kernel 1: x (b, i, n) f32  ->  Xh (b, n, i) f16 ------------
__global__ __launch_bounds__(256) void transpose_x(const float* x, _Float16* Xh) {
  int tid = blockIdx.x * 256 + threadIdx.x;          // (b, n, i) order
  int i = tid % DIMX;
  int n = (tid / DIMX) % NNX;
  int b = tid / (DIMX * NNX);
  Xh[tid] = (_Float16)(x[(b * DIMX + i) * NNX + n]);
}

// ---------------- kernel 2: QKV GEMM (O=1536, N=784, K=384) via WMMA ---------
__global__ __launch_bounds__(256) void qkv_gemm(
    const _Float16* Wh, const float* biasF, const _Float16* Xh,
    _Float16* qh, _Float16* kh_t, float* vf32, _Float16* vh16) {
  int wave = threadIdx.x >> 5, lane = threadIdx.x & 31;
  int tile = blockIdx.x * 8 + wave;                  // 16*96*49 tiles
  int nt = tile % 49, rem = tile / 49;
  int ot = rem % 96, b = rem / 96;
  int o0 = ot * 16, n0 = nt * 16;
  const _Float16* Arow = Wh + o0 * DIMX;
  const _Float16* Brow = Xh + (b * NNX + n0) * DIMX;
  v8f c = zero8();
#pragma unroll
  for (int k0 = 0; k0 < DIMX; k0 += 32) {
    v16h a  = load_frag(Arow, DIMX, lane, k0);
    v16h bb = load_frag(Brow, DIMX, lane, k0);
    c = wmma16(a, bb, c);
  }
  int col = lane & 15, rh = (lane >> 4) * 8;
  int n = n0 + col;
#pragma unroll
  for (int j = 0; j < 8; ++j) {
    int o = o0 + j + rh;
    float val = c[j] + biasF[o];
    if (o < 512) {
      _Float16 hv = (_Float16)val;
      if (o < 256) qh[(b * NNX + n) * HKX + o] = hv;
      else         kh_t[(b * NPAD + n) * HKX + (o - 256)] = hv;
    } else {
      int ch = o - 512;
      vf32[(b * DHX + ch) * NNX + n] = val;
      vh16[(b * DHX + ch) * NPAD + n] = (_Float16)val;
    }
  }
}

// ---------------- kernel 3: Qp[b,h,n,c] = SCALE * th1[h, c/32] * q[b,n,c] ----
__global__ __launch_bounds__(256) void expand_qp(const _Float16* qh, const float* th1_w, _Float16* Qp) {
  int tid = blockIdx.x * 256 + threadIdx.x;          // 16*8*784*256
  int c = tid & 255;
  int n = (tid >> 8) % NNX;
  int h = (tid / (HKX * NNX)) % NHX;
  int b = tid / (HKX * NNX * NHX);
  float v = (float)qh[(b * NNX + n) * HKX + c];
  Qp[tid] = (_Float16)(SCALEX * th1_w[h * NHX + (c >> 5)] * v);
}

// ---------------- kernel 4: zero pad regions (m = 784..799) ------------------
__global__ __launch_bounds__(256) void pad_zero(_Float16* kh_t, _Float16* vh16) {
  int tid = blockIdx.x * 256 + threadIdx.x;
  if (tid < BB * 16 * HKX) {
    int c = tid & 255, r = (tid >> 8) & 15, b = tid >> 12;
    kh_t[(b * NPAD + NNX + r) * HKX + c] = (_Float16)0.f;
  } else {
    int t = tid - BB * 16 * HKX;                     // 16*1024*16
    int mm = t & 15, ch = (t >> 4) & 1023, b = t >> 14;
    vh16[(b * DHX + ch) * NPAD + NNX + mm] = (_Float16)0.f;
  }
}

// ---------------- kernel 5: Vsum[b,h,d] = sum_m v ----------------------------
__global__ __launch_bounds__(256) void vsum_kernel(const _Float16* vh16, float* Vsum) {
  int tid = blockIdx.x * 256 + threadIdx.x;          // 16*8*128
  int d = tid & 127, h = (tid >> 7) & 7, b = tid >> 10;
  const _Float16* row = vh16 + (size_t)(b * DHX + h * DDX + d) * NPAD;
  float s = 0.f;
  for (int m = 0; m < NNX; ++m) s += (float)row[m];
  Vsum[tid] = s;
}

// ---------------- kernel 6: fused attention ---------------------------------
// block = 8 waves; wave h owns head h for one (b, 16-row) tile.
__global__ __launch_bounds__(256) void attn_kernel(
    const _Float16* Qp, const _Float16* kh_t, const _Float16* vh16,
    const float* tabMix, const float* th1_b, const float* th2_w, const float* th2_b,
    const float* Vsum, const int* bias_idx, float* A_out) {
  __shared__ _Float16 p_lds[NHX * 16 * 32];          // 8 KB: per-head P tiles
  int wave = threadIdx.x >> 5, lane = threadIdx.x & 31;
  int b = blockIdx.x / 49, nt = blockIdx.x % 49;
  int n0 = nt * 16, h = wave;
  int col = lane & 15, rh = (lane >> 4) * 8;

  // resident Q' fragments (th1 & scale folded): 16 rows x 256 K
  const _Float16* Qbase = Qp + ((size_t)(b * NHX + h) * NNX + n0) * HKX;
  v16h qa[8];
#pragma unroll
  for (int kk = 0; kk < 8; ++kk) qa[kk] = load_frag(Qbase, HKX, lane, kk * 32);

  float th1b = th1_b[h];
  float run_m[8], run_s[8];
#pragma unroll
  for (int j = 0; j < 8; ++j) { run_m[j] = -1e30f; run_s[j] = 0.f; }

  // ---- pass 1: streaming max/sum over all 49 m-tiles ----
  for (int mt = 0; mt < 49; ++mt) {
    int m0 = mt * 16;
    const _Float16* Bbase = kh_t + ((size_t)b * NPAD + m0) * HKX;
    v8f c = zero8();
#pragma unroll
    for (int kk = 0; kk < 8; ++kk) c = wmma16(qa[kk], load_frag(Bbase, HKX, lane, kk * 32), c);
    int m = m0 + col;
#pragma unroll
    for (int j = 0; j < 8; ++j) {
      int n = n0 + j + rh;
      int t = bias_idx[n * NNX + m];
      float l = c[j] + tabMix[h * NNX + t] + th1b;
      float rm = l;
      rm = fmaxf(rm, __shfl_xor(rm, 1));
      rm = fmaxf(rm, __shfl_xor(rm, 2));
      rm = fmaxf(rm, __shfl_xor(rm, 4));
      rm = fmaxf(rm, __shfl_xor(rm, 8));
      float nm = fmaxf(run_m[j], rm);
      float pe = __expf(l - nm);
      float rs = pe;
      rs += __shfl_xor(rs, 1);
      rs += __shfl_xor(rs, 2);
      rs += __shfl_xor(rs, 4);
      rs += __shfl_xor(rs, 8);
      run_s[j] = run_s[j] * __expf(run_m[j] - nm) + rs;
      run_m[j] = nm;
    }
  }
  float inv_s[8];
#pragma unroll
  for (int j = 0; j < 8; ++j) inv_s[j] = 1.f / run_s[j];

  // ---- pass 2: recompute, normalize, th2-mix via LDS, P2 @ V ----
  v8f outc[8];
#pragma unroll
  for (int dt = 0; dt < 8; ++dt) outc[dt] = zero8();
  float th2r[8];
#pragma unroll
  for (int h2 = 0; h2 < NHX; ++h2) th2r[h2] = th2_w[h * NHX + h2];

  for (int ms = 0; ms < 25; ++ms) {                  // 25 * 32 = 800 (padded)
    int m0 = ms * 32;
#pragma unroll
    for (int half = 0; half < 2; ++half) {
      int mh0 = m0 + half * 16;
      const _Float16* Bbase = kh_t + ((size_t)b * NPAD + mh0) * HKX;
      v8f c = zero8();
#pragma unroll
      for (int kk = 0; kk < 8; ++kk) c = wmma16(qa[kk], load_frag(Bbase, HKX, lane, kk * 32), c);
      int m = mh0 + col;
#pragma unroll
      for (int j = 0; j < 8; ++j) {
        float p = 0.f;
        if (m < NNX) {
          int n = n0 + j + rh;
          int t = bias_idx[n * NNX + m];
          float l = c[j] + tabMix[h * NNX + t] + th1b;
          p = __expf(l - run_m[j]) * inv_s[j];
        }
        p_lds[(h * 16 + j + rh) * 32 + half * 16 + col] = (_Float16)p;
      }
    }
    __syncthreads();
    // build th2-mixed A fragment (16n x 32m) from all 8 heads' P tiles
    v16h pa;
    int nl = lane & 15;
    int mb = (lane >> 4) * 8;
#pragma unroll
    for (int s = 0; s < 16; ++s) {
      int mm = mb + (s & 7) + ((s >> 3) << 4);
      float acc = 0.f;
#pragma unroll
      for (int h2 = 0; h2 < NHX; ++h2) acc += th2r[h2] * (float)p_lds[(h2 * 16 + nl) * 32 + mm];
      pa[s] = (_Float16)acc;
    }
    __syncthreads();
    // outc += P2 (16x32) @ V (32m x 16d), 8 d-tiles
    const _Float16* Vbase = vh16 + ((size_t)(b * DHX) + h * DDX) * NPAD + m0;
#pragma unroll
    for (int dt = 0; dt < 8; ++dt) {
      v16h vb = load_frag(Vbase + dt * 16 * NPAD, NPAD, lane, 0);
      outc[dt] = wmma16(pa, vb, outc[dt]);
    }
  }

  // epilogue: + th2_b * Vsum ; write (b, n, c=h*128+d) f32
  float t2b = th2_b[h];
#pragma unroll
  for (int dt = 0; dt < 8; ++dt) {
    int d = dt * 16 + col;
    float vs = t2b * Vsum[(b * NHX + h) * DDX + d];
#pragma unroll
    for (int j = 0; j < 8; ++j) {
      int n = n0 + j + rh;
      A_out[((size_t)b * NNX + n) * DHX + h * DDX + d] = outc[dt][j] + vs;
    }
  }
}

// ---------------- kernel 7: v_local dwconv3 + BN + add + ReLU -> ah f16 -----
__global__ __launch_bounds__(256) void combine_kernel(
    const float* vf32, const float* A_out,
    const float* vl_w, const float* vl_b, const float* vl_g, const float* vl_be,
    const float* vl_m, const float* vl_rv, _Float16* ah) {
  int tid = blockIdx.x * 256 + threadIdx.x;          // (b, c, n) order
  int n = tid % NNX;
  int c = (tid / NNX) % DHX;
  int b = tid / (NNX * DHX);
  int r = n / RRX, cc = n % RRX;
  const float* in = vf32 + (size_t)(b * DHX + c) * NNX;
  const float* w = vl_w + c * 9;
  float acc = 0.f;
#pragma unroll
  for (int dy = -1; dy <= 1; ++dy) {
#pragma unroll
    for (int dx = -1; dx <= 1; ++dx) {
      int r2 = r + dy, c2 = cc + dx;
      if (r2 >= 0 && r2 < RRX && c2 >= 0 && c2 < RRX)
        acc += w[(dy + 1) * 3 + (dx + 1)] * in[r2 * RRX + c2];
    }
  }
  acc += vl_b[c];
  float inv = vl_g[c] * rsqrtf(vl_rv[c] + EPSX);
  float vloc = acc * inv + (vl_be[c] - vl_m[c] * inv);
  float a = A_out[((size_t)b * NNX + n) * DHX + c];
  float val = fmaxf(a + vloc, 0.f);
  ah[((size_t)b * NNX + n) * DHX + c] = (_Float16)val;
}

// ---------------- kernel 8: proj GEMM (O=384, N=784, K=1024) ----------------
__global__ __launch_bounds__(256) void proj_gemm(
    const _Float16* Wph, const float* biasP, const _Float16* ah, float* out) {
  int wave = threadIdx.x >> 5, lane = threadIdx.x & 31;
  int tile = blockIdx.x * 8 + wave;                  // 16*24*49 tiles
  int nt = tile % 49, rem = tile / 49;
  int ot = rem % 24, b = rem / 24;
  int o0 = ot * 16, n0 = nt * 16;
  const _Float16* Arow = Wph + o0 * DHX;
  const _Float16* Brow = ah + ((size_t)b * NNX + n0) * DHX;
  v8f c = zero8();
  for (int k0 = 0; k0 < DHX; k0 += 32) {
    v16h a  = load_frag(Arow, DHX, lane, k0);
    v16h bb = load_frag(Brow, DHX, lane, k0);
    c = wmma16(a, bb, c);
  }
  int col = lane & 15, rh = (lane >> 4) * 8;
  int n = n0 + col;
#pragma unroll
  for (int j = 0; j < 8; ++j) {
    int o = o0 + j + rh;
    out[((size_t)b * DIMX + o) * NNX + n] = c[j] + biasP[o];
  }
}

// ---------------- launcher ---------------------------------------------------
extern "C" void kernel_launch(void* const* d_in, const int* in_sizes, int n_in,
                              void* d_out, int out_size, void* d_ws, size_t ws_size,
                              hipStream_t stream) {
  const float* x     = (const float*)d_in[0];
  const float* q_w   = (const float*)d_in[1];
  const float* q_b   = (const float*)d_in[2];
  const float* q_g   = (const float*)d_in[3];
  const float* q_be  = (const float*)d_in[4];
  const float* q_m   = (const float*)d_in[5];
  const float* q_rv  = (const float*)d_in[6];
  const float* k_w   = (const float*)d_in[7];
  const float* k_b   = (const float*)d_in[8];
  const float* k_g   = (const float*)d_in[9];
  const float* k_be  = (const float*)d_in[10];
  const float* k_m   = (const float*)d_in[11];
  const float* k_rv  = (const float*)d_in[12];
  const float* v_w   = (const float*)d_in[13];
  const float* v_b   = (const float*)d_in[14];
  const float* v_g   = (const float*)d_in[15];
  const float* v_be  = (const float*)d_in[16];
  const float* v_m   = (const float*)d_in[17];
  const float* v_rv  = (const float*)d_in[18];
  const float* vl_w  = (const float*)d_in[19];
  const float* vl_b  = (const float*)d_in[20];
  const float* vl_g  = (const float*)d_in[21];
  const float* vl_be = (const float*)d_in[22];
  const float* vl_m  = (const float*)d_in[23];
  const float* vl_rv = (const float*)d_in[24];
  const float* th1_w = (const float*)d_in[25];
  const float* th1_b = (const float*)d_in[26];
  const float* th2_w = (const float*)d_in[27];
  const float* th2_b = (const float*)d_in[28];
  const float* p_w   = (const float*)d_in[29];
  const float* p_b   = (const float*)d_in[30];
  const float* p_g   = (const float*)d_in[31];
  const float* p_be  = (const float*)d_in[32];
  const float* p_m   = (const float*)d_in[33];
  const float* p_rv  = (const float*)d_in[34];
  const float* btab  = (const float*)d_in[35];
  const int*   bidx  = (const int*)d_in[36];
  float* out = (float*)d_out;

  char* ws = (char*)d_ws;
  size_t off = 0;
  auto take = [&](size_t bytes) -> char* {
    off = (off + 255) & ~(size_t)255;
    char* p = ws + off;
    off += bytes;
    return p;
  };
  _Float16* Wh    = (_Float16*)take((size_t)1536 * DIMX * 2);
  float*    biasF = (float*)take(1536 * 4);
  _Float16* Wph   = (_Float16*)take((size_t)DIMX * DHX * 2);
  float*    biasP = (float*)take(DIMX * 4);
  float*    tabMix= (float*)take((size_t)NHX * NNX * 4);
  _Float16* Xh    = (_Float16*)take((size_t)BB * NNX * DIMX * 2);
  _Float16* qh    = (_Float16*)take((size_t)BB * NNX * HKX * 2);
  _Float16* kh_t  = (_Float16*)take((size_t)BB * NPAD * HKX * 2);
  _Float16* vh16  = (_Float16*)take((size_t)BB * DHX * NPAD * 2);
  float*    vf32  = (float*)take((size_t)BB * DHX * NNX * 4);
  _Float16* Qp    = (_Float16*)take((size_t)BB * NHX * NNX * HKX * 2);
  float*    Vsum  = (float*)take((size_t)BB * NHX * DDX * 4);
  float*    A_out = (float*)take((size_t)BB * NNX * DHX * 4);
  _Float16* ah    = (_Float16*)take((size_t)BB * NNX * DHX * 2);
  (void)ws_size; (void)n_in; (void)in_sizes; (void)out_size;

  {
    int total = 1536 * DIMX + DIMX * DHX + 1536 + DIMX + NHX * NNX;
    prep_params<<<(total + 255) / 256, 256, 0, stream>>>(
        q_w, q_b, q_g, q_be, q_m, q_rv, k_w, k_b, k_g, k_be, k_m, k_rv,
        v_w, v_b, v_g, v_be, v_m, v_rv, p_w, p_b, p_g, p_be, p_m, p_rv,
        th1_w, btab, Wh, biasF, Wph, biasP, tabMix);
  }
  transpose_x<<<(BB * NNX * DIMX) / 256, 256, 0, stream>>>(x, Xh);
  qkv_gemm<<<(16 * 96 * 49) / 8, 256, 0, stream>>>(Wh, biasF, Xh, qh, kh_t, vf32, vh16);
  expand_qp<<<(BB * NHX * NNX * HKX) / 256, 256, 0, stream>>>(qh, th1_w, Qp);
  pad_zero<<<(BB * 16 * HKX + BB * DHX * 16) / 256, 256, 0, stream>>>(kh_t, vh16);
  vsum_kernel<<<(BB * NHX * DDX) / 256, 256, 0, stream>>>(vh16, Vsum);
  attn_kernel<<<BB * 49, 256, 0, stream>>>(Qp, kh_t, vh16, tabMix, th1_b, th2_w, th2_b,
                                           Vsum, bidx, A_out);
  combine_kernel<<<(BB * DHX * NNX) / 256, 256, 0, stream>>>(
      vf32, A_out, vl_w, vl_b, vl_g, vl_be, vl_m, vl_rv, ah);
  proj_gemm<<<(16 * 24 * 49) / 8, 256, 0, stream>>>(Wph, biasP, ah, out);
}
